// BertFlashSelfAttention_40896678592933
// MI455X (gfx1250) — matrix operand order repaired and checked by reading, not compiled, not run on hardware
//
#include <hip/hip_runtime.h>
#include <cstdint>
#include <cstddef>

typedef __bf16 bf16_t;
typedef __attribute__((ext_vector_type(16))) __bf16 v16bf;
typedef __attribute__((ext_vector_type(8)))  float  v8f;

union Frag16 { v16bf v; uint4 u[2]; };

static __device__ __forceinline__ v8f wmma_bf16(Frag16 a, Frag16 b, v8f c) {
  // D = A(16x32 bf16) x B(32x16 bf16) + C(16x16 f32)
  return __builtin_amdgcn_wmma_f32_16x16x32_bf16(false, a.v, false, b.v,
                                                 (short)0, c, false, false);
}

// Async copy 16B from global to LDS (gfx1250 ASYNCcnt path).
static __device__ __forceinline__ void async_g2l_b128(uint32_t lds_off,
                                                      const void* gaddr) {
  asm volatile("global_load_async_to_lds_b128 %0, %1, off"
               :: "v"(lds_off), "v"((unsigned long long)(uintptr_t)gaddr)
               : "memory");
}
static __device__ __forceinline__ void wait_async0() {
  asm volatile("s_wait_asynccnt 0x0" ::: "memory");
}

// ---------------- f32 -> bf16 convert ----------------
__global__ __launch_bounds__(256) void cvt_f32_bf16(const float* __restrict__ in,
                                                    bf16_t* __restrict__ out,
                                                    long long n) {
  long long i = (long long)blockIdx.x * blockDim.x + threadIdx.x;
  long long stride = (long long)gridDim.x * blockDim.x;
  for (; i < n; i += stride) out[i] = (bf16_t)in[i];
}

// ---------------- QKV projection GEMM: C(MxN) = A(MxK) * W(NxK)^T + bias ----
// Block = 256 threads = 8 waves (2x4 wave grid), 128x128 tile; wave = 64x32.
__global__ __launch_bounds__(256) void qkv_gemm(const bf16_t* __restrict__ A,
                                                const bf16_t* __restrict__ W,
                                                const float* __restrict__ bias,
                                                bf16_t* __restrict__ C,
                                                int M, int N, int K) {
  const int wave = threadIdx.x >> 5;
  const int lane = threadIdx.x & 31;
  const int half = lane >> 4;
  const int l16  = lane & 15;
  const int mBase = blockIdx.x * 128 + (wave >> 2) * 64;
  const int nBase = blockIdx.y * 128 + (wave & 3) * 32;

  v8f acc[4][2];
#pragma unroll
  for (int mt = 0; mt < 4; ++mt)
#pragma unroll
    for (int nt = 0; nt < 2; ++nt)
#pragma unroll
      for (int e = 0; e < 8; ++e) acc[mt][nt][e] = 0.f;

  for (int kk = 0; kk < K; kk += 32) {
    Frag16 af[4];
#pragma unroll
    for (int mt = 0; mt < 4; ++mt) {
      // A-frag: lanes 0-15 hold K [0..7]+[16..23], lanes 16-31 hold [8..15]+[24..31]
      const bf16_t* p = A + (size_t)(mBase + mt * 16 + l16) * K + kk + half * 8;
      af[mt].u[0] = *(const uint4*)p;
      af[mt].u[1] = *(const uint4*)(p + 16);
    }
    Frag16 bfr[2];
#pragma unroll
    for (int nt = 0; nt < 2; ++nt) {
      // B-frag: B[k][n] = W[n][k]; lanes 0-15 hold K [0..7]+[16..23], lanes 16-31 hold [8..15]+[24..31]
      const bf16_t* p = W + (size_t)(nBase + nt * 16 + l16) * K + kk + half * 8;
      bfr[nt].u[0] = *(const uint4*)p;
      bfr[nt].u[1] = *(const uint4*)(p + 16);
    }
#pragma unroll
    for (int mt = 0; mt < 4; ++mt)
#pragma unroll
      for (int nt = 0; nt < 2; ++nt)
        acc[mt][nt] = wmma_bf16(af[mt], bfr[nt], acc[mt][nt]);
  }

  float bv[2];
#pragma unroll
  for (int nt = 0; nt < 2; ++nt) bv[nt] = bias[nBase + nt * 16 + l16];

#pragma unroll
  for (int mt = 0; mt < 4; ++mt)
#pragma unroll
    for (int nt = 0; nt < 2; ++nt)
#pragma unroll
      for (int r = 0; r < 8; ++r) {
        // C-layout: lane col = l16, row = r + 8*half
        int row = mBase + mt * 16 + r + half * 8;
        int col = nBase + nt * 16 + l16;
        C[(size_t)row * N + col] = (bf16_t)(acc[mt][nt][r] + bv[nt]);
      }
}

// ---------------- V transpose: vt[h*64+d][token] = qkv[token][2*dim + h*64+d]
__global__ __launch_bounds__(256) void transpose_v(const bf16_t* __restrict__ qkv,
                                                   bf16_t* __restrict__ vt,
                                                   int nnz, int dim) {
  __shared__ bf16_t tile[64][65];
  const int tb = blockIdx.x * 64;   // token base
  const int db = blockIdx.y * 64;   // (head,dim) base
  const int c  = threadIdx.x & 63;
  const int r0 = threadIdx.x >> 6;  // 0..3
#pragma unroll
  for (int i = 0; i < 16; ++i) {
    int r = r0 + i * 4;
    tile[r][c] = qkv[(size_t)(tb + r) * (3 * dim) + 2 * dim + db + c];
  }
  __syncthreads();
#pragma unroll
  for (int i = 0; i < 16; ++i) {
    int r = r0 + i * 4;
    vt[(size_t)(db + r) * nnz + tb + c] = tile[c][r];
  }
}

// ---------------- varlen flash attention, one (64-query block, head) per CTA
// 4 waves x 16 query rows; K/V tiles async-staged to LDS once per key block.
__global__ __launch_bounds__(128) void flash_attn(const bf16_t* __restrict__ qkv,
                                                  const bf16_t* __restrict__ vt,
                                                  const int* __restrict__ cu,
                                                  float* __restrict__ out,
                                                  int nnz, int dim, int ncu) {
  const int h    = blockIdx.y;
  const int row0 = blockIdx.x * 64;      // seqlens are multiples of 64: no straddle
  int b = 0;
  while (b + 2 < ncu && cu[b + 1] <= row0) ++b;
  const int k0 = cu[b], k1 = cu[b + 1];

  const int tid  = threadIdx.x;
  const int wave = tid >> 5;
  const int lane = tid & 31;
  const int half = lane >> 4;
  const int l16  = lane & 15;
  const int qrow = row0 + wave * 16;
  const int N3   = 3 * dim;

  __shared__ __align__(16) bf16_t k_lds[64][72];     // keys x head-dim
  __shared__ __align__(16) bf16_t v_lds[64][72];     // head-dim x keys (from vt)
  __shared__ __align__(16) bf16_t p_lds[4][16][72];  // per-wave P tile

  // Q fragments (16 rows x 64 head-dims), loaded once
  Frag16 qf[2];
  {
    const bf16_t* qp = qkv + (size_t)(qrow + l16) * N3 + h * 64;
#pragma unroll
    for (int st = 0; st < 2; ++st) {
      const bf16_t* p = qp + st * 32 + half * 8;
      qf[st].u[0] = *(const uint4*)p;
      qf[st].u[1] = *(const uint4*)(p + 16);
    }
  }

  v8f o[4];
#pragma unroll
  for (int t = 0; t < 4; ++t)
#pragma unroll
    for (int e = 0; e < 8; ++e) o[t][e] = 0.f;
  float rm[8], rl[8];
#pragma unroll
  for (int r = 0; r < 8; ++r) { rm[r] = -1e30f; rl[r] = 0.f; }

  for (int kb = k0; kb < k1; kb += 64) {
    // ---- async-stage K tile (64x64) and V tile (64x64) into LDS ----
    // 512 b128 chunks per tile; 128 threads x 4 chunks each per tile.
#pragma unroll
    for (int i = 0; i < 4; ++i) {
      int c    = tid + i * 128;       // 0..511
      int rowc = c >> 3;              // 0..63
      int col  = (c & 7) * 8;         // 0..56, in elements
      const bf16_t* kg = qkv + (size_t)(kb + rowc) * N3 + dim + h * 64 + col;
      async_g2l_b128((uint32_t)(uintptr_t)&k_lds[rowc][col], kg);
      const bf16_t* vg = vt + (size_t)(h * 64 + rowc) * nnz + kb + col;
      async_g2l_b128((uint32_t)(uintptr_t)&v_lds[rowc][col], vg);
    }
    wait_async0();
    __syncthreads();

    // ---- S = Q K^T (16 x 64 tile, 8 WMMAs), K frags from LDS ----
    v8f s[4];
#pragma unroll
    for (int j = 0; j < 4; ++j) {
#pragma unroll
      for (int e = 0; e < 8; ++e) s[j][e] = 0.f;
#pragma unroll
      for (int st = 0; st < 2; ++st) {
        Frag16 kf;  // B[k=d][n=key]: d [0..7]+[16..23] / [8..15]+[24..31] per half
        const bf16_t* p = &k_lds[j * 16 + l16][st * 32 + half * 8];
        kf.u[0] = *(const uint4*)p;
        kf.u[1] = *(const uint4*)(p + 16);
        s[j] = wmma_bf16(qf[st], kf, s[j]);
      }
    }

    // ---- online softmax (row = r + 8*half lives across one 16-lane half) ----
    const float scale = 0.125f;  // 1/sqrt(64)
#pragma unroll
    for (int r = 0; r < 8; ++r) {
      float mx = -1e30f;
#pragma unroll
      for (int j = 0; j < 4; ++j) { s[j][r] *= scale; mx = fmaxf(mx, s[j][r]); }
#pragma unroll
      for (int off = 8; off > 0; off >>= 1) mx = fmaxf(mx, __shfl_xor(mx, off, 32));
      float nm    = fmaxf(rm[r], mx);
      float alpha = __expf(rm[r] - nm);
      float rs = 0.f;
#pragma unroll
      for (int j = 0; j < 4; ++j) {
        float pv = __expf(s[j][r] - nm);
        s[j][r] = pv;
        rs += pv;
      }
#pragma unroll
      for (int off = 8; off > 0; off >>= 1) rs += __shfl_xor(rs, off, 32);
      rl[r] = rl[r] * alpha + rs;
      rm[r] = nm;
#pragma unroll
      for (int t = 0; t < 4; ++t) o[t][r] *= alpha;
    }

    // ---- C-layout P -> LDS -> A-layout P (same-wave DS ordering) ----
#pragma unroll
    for (int j = 0; j < 4; ++j)
#pragma unroll
      for (int r = 0; r < 8; ++r)
        p_lds[wave][r + half * 8][j * 16 + l16] = (bf16_t)s[j][r];

    // ---- O += P V (8 WMMAs), V frags from LDS ----
#pragma unroll
    for (int st = 0; st < 2; ++st) {
      Frag16 pf;
      const bf16_t* pp = &p_lds[wave][l16][st * 32 + half * 8];
      pf.u[0] = *(const uint4*)pp;
      pf.u[1] = *(const uint4*)(pp + 16);
#pragma unroll
      for (int t = 0; t < 4; ++t) {
        Frag16 vf;  // B[k=key][n=d]: key [0..7]+[16..23] / [8..15]+[24..31] per half (transposed V)
        const bf16_t* vp = &v_lds[t * 16 + l16][st * 32 + half * 8];
        vf.u[0] = *(const uint4*)vp;
        vf.u[1] = *(const uint4*)(vp + 16);
        o[t] = wmma_bf16(pf, vf, o[t]);
      }
    }
    __syncthreads();  // protect k_lds/v_lds before next async stage
  }

  // ---- normalize + store f32 output ----
#pragma unroll
  for (int r = 0; r < 8; ++r) {
    float inv = 1.0f / rl[r];
    int m = qrow + r + half * 8;
#pragma unroll
    for (int t = 0; t < 4; ++t)
      out[(size_t)m * dim + h * 64 + t * 16 + l16] = o[t][r] * inv;
  }
}

extern "C" void kernel_launch(void* const* d_in, const int* in_sizes, int n_in,
                              void* d_out, int out_size, void* d_ws, size_t ws_size,
                              hipStream_t stream) {
  const float* hidden = (const float*)d_in[0];
  const float* Wq     = (const float*)d_in[1];
  const float* bias   = (const float*)d_in[2];
  const int*   cu     = (const int*)d_in[3];
  float*       out    = (float*)d_out;

  const int N3  = in_sizes[2];        // 3072
  const int dim = N3 / 3;             // 1024
  const int nnz = in_sizes[0] / dim;  // 8704
  const int ncu = in_sizes[3];        // 17

  auto align256 = [](size_t x) { return (x + 255) & ~(size_t)255; };
  char* ws = (char*)d_ws;
  size_t off = 0;
  bf16_t* hid_bf = (bf16_t*)(ws + off); off += align256((size_t)nnz * dim * sizeof(bf16_t));
  bf16_t* w_bf   = (bf16_t*)(ws + off); off += align256((size_t)N3 * dim * sizeof(bf16_t));
  bf16_t* qkv_bf = (bf16_t*)(ws + off); off += align256((size_t)nnz * N3 * sizeof(bf16_t));
  bf16_t* vt_bf  = (bf16_t*)(ws + off); off += align256((size_t)dim * nnz * sizeof(bf16_t));
  (void)ws_size; (void)n_in; (void)out_size;

  cvt_f32_bf16<<<2048, 256, 0, stream>>>(hidden, hid_bf, (long long)nnz * dim);
  cvt_f32_bf16<<<2048, 256, 0, stream>>>(Wq, w_bf, (long long)N3 * dim);

  qkv_gemm<<<dim3(nnz / 128, N3 / 128), 256, 0, stream>>>(hid_bf, w_bf, bias,
                                                          qkv_bf, nnz, N3, dim);
  transpose_v<<<dim3(nnz / 64, dim / 64), 256, 0, stream>>>(qkv_bf, vt_bf, nnz, dim);
  flash_attn<<<dim3(nnz / 64, dim / 64), 128, 0, stream>>>(qkv_bf, vt_bf, cu, out,
                                                           nnz, dim, ncu);
}
